// Seq_Transformer_64622077935836
// MI455X (gfx1250) — matrix operand order repaired and checked by reading, not compiled, not run on hardware
//
#include <hip/hip_runtime.h>
#include <math.h>

// ---------------------------------------------------------------------------
// CDNA5 (gfx1250) implementation: wave32 + v_wmma_f32_16x16x32_bf16 GEMM core
// used for every conv (implicit GEMM), transposed conv, and linear layer.
// Block tile 128x64xK64, 8 waves; each wave computes a 32x32 D slab and
// issues 8 WMMAs per staging round (two 32-deep K sub-tiles).
// ---------------------------------------------------------------------------

typedef __attribute__((ext_vector_type(16))) __bf16 v16bf;
typedef __attribute__((ext_vector_type(8)))  float v8f;
static_assert(sizeof(v16bf) == 32, "v16bf must be 8 dwords");

__device__ __forceinline__ unsigned short f2bf(float f) {
  unsigned u = __builtin_bit_cast(unsigned, f);
  unsigned r = u + 0x7FFFu + ((u >> 16) & 1u);   // round-to-nearest-even
  return (unsigned short)(r >> 16);
}
__device__ __forceinline__ unsigned pack2bf(float a, float b) {
  return (unsigned)f2bf(a) | ((unsigned)f2bf(b) << 16);
}
__device__ __forceinline__ float sigm(float x) { return 1.0f / (1.0f + expf(-x)); }
__device__ __forceinline__ unsigned pcg(unsigned x) {
  x = x * 747796405u + 2891336453u;
  unsigned w = ((x >> ((x >> 28) + 4u)) ^ x) * 277803737u;
  return (w >> 22) ^ w;
}
#define LOG2PI_F 1.8378770664093453f

// ------------------------------ WMMA GEMM ----------------------------------
// C[M,N] = act( A' @ B'^T + bias ), bf16 inputs, f32 accumulate.
// MODE 0: A dense (M,K); B dense (N,K).                      (x @ W^T)
// MODE 1: A = im2col(conv k3 p1 stride s) of NCHW input; B = (Co, Ci*9) dense.
//         C scattered to NCHW (BS,Co,Ho,Wo).
// MODE 2: A = im2col of ConvTranspose2d(k3,s2,p1,op1); B gathered from
//         w(Ci,Co,3,3) flipped+transposed. C scattered NCHW.
// MODE 3: A = concat(A, A2) along K (each M x K/2 dense); B dense.
struct GemmP {
  const float* A; const float* A2; const float* B; const float* bias; float* C;
  int M, N, K, act;                 // act: 0 none, 1 relu, 2 leaky(0.01)
  int Ci, H, W, Ho, Wo, stride, Co; // conv modes only
};

#define BM 128
#define BN 64
#define BKK 64
#define LDT 66   // LDS row stride in ushorts (pad to dodge bank conflicts)

template <int MODE>
__global__ __launch_bounds__(256) void k_gemm(GemmP p) {
  __shared__ unsigned short As[BM * LDT];
  __shared__ unsigned short Bs[BN * LDT];
  const int tid   = threadIdx.x;
  const int tileM = blockIdx.y * BM;
  const int tileN = blockIdx.x * BN;
  const int lane  = tid & 31;
  const int wave  = tid >> 5;
  const int wm    = wave >> 1;   // 0..3 : 32-row slab
  const int wn    = wave & 1;    // 0..1 : 32-col slab
  const int HoWo  = p.Ho * p.Wo;
  const int Khalf = p.K >> 1;    // MODE 3 only

  // staging coords: A -> 128 rows x 64 cols, 32 elems/thread (one half-row)
  const int ar = tid >> 1;              // 0..127
  const int ac = (tid & 1) * 32;        // 0 | 32
  // staging coords: B -> 64 rows x 64 cols, 16 elems/thread
  const int br = tid >> 2;              // 0..63
  const int bc = (tid & 3) * 16;        // 0,16,32,48

  v8f acc00 = {}, acc01 = {}, acc10 = {}, acc11 = {};

  for (int k0 = 0; k0 < p.K; k0 += BKK) {
    const bool fullK = (k0 + BKK) <= p.K;
    // ---- stage A tile (BM x BKK) as bf16 into LDS ----
    {
      const int gm = tileM + ar;
      unsigned* dst = (unsigned*)&As[ar * LDT + ac];
      if ((MODE == 0 || MODE == 3) && gm < p.M && fullK) {
        const float* src;
        if (MODE == 3) {
          const int gk = k0 + ac;   // tile never straddles the concat seam
          src = (gk < Khalf) ? (p.A + (size_t)gm * Khalf + gk)
                             : (p.A2 + (size_t)gm * Khalf + (gk - Khalf));
        } else {
          src = p.A + (size_t)gm * p.K + k0 + ac;
        }
        #pragma unroll
        for (int j = 0; j < 8; ++j) {
          const float4 f = *(const float4*)(src + 4 * j);
          dst[2 * j]     = pack2bf(f.x, f.y);
          dst[2 * j + 1] = pack2bf(f.z, f.w);
        }
        if (MODE == 0 && (k0 + BKK) < p.K)
          __builtin_prefetch(src + BKK, 0, 0);
      } else if (MODE == 0 || MODE == 3) {      // edge tile (rare)
        #pragma unroll
        for (int j = 0; j < 16; ++j) {
          const int gk0 = k0 + ac + 2 * j, gk1 = gk0 + 1;
          float x = 0.0f, y = 0.0f;
          if (gm < p.M) {
            if (MODE == 0) {
              if (gk0 < p.K) x = p.A[(size_t)gm * p.K + gk0];
              if (gk1 < p.K) y = p.A[(size_t)gm * p.K + gk1];
            } else {
              if (gk0 < p.K) x = (gk0 < Khalf) ? p.A[(size_t)gm * Khalf + gk0]
                                               : p.A2[(size_t)gm * Khalf + gk0 - Khalf];
              if (gk1 < p.K) y = (gk1 < Khalf) ? p.A[(size_t)gm * Khalf + gk1]
                                               : p.A2[(size_t)gm * Khalf + gk1 - Khalf];
            }
          }
          dst[j] = pack2bf(x, y);
        }
      } else {
        // conv / convT im2col gather: branchless clamped-index loads
        const int nb = gm / HoWo, rem = gm - nb * HoWo;
        const int ho = rem / p.Wo, wo = rem - ho * p.Wo;
        const bool mok = gm < p.M;
        float vv[32];
        #pragma unroll
        for (int j = 0; j < 32; ++j) {
          const int gk = k0 + ac + j;
          const int ci = gk / 9, r9 = gk - ci * 9;
          const int kh = r9 / 3, kw = r9 - kh * 3;
          int hi, wi;
          bool ok = mok && (gk < p.K);
          if (MODE == 1) {
            hi = ho * p.stride - 1 + kh;
            wi = wo * p.stride - 1 + kw;
            ok = ok && hi >= 0 && hi < p.H && wi >= 0 && wi < p.W;
          } else {
            const int ph = ho + kh - 1, pw = wo + kw - 1;
            hi = ph >> 1; wi = pw >> 1;
            ok = ok && ph >= 0 && pw >= 0 && !(ph & 1) && !(pw & 1) && hi < p.H && wi < p.W;
          }
          const size_t idx = ok ? ((((size_t)nb * p.Ci + ci) * p.H + hi) * p.W + wi) : 0;
          const float t = p.A[idx];
          vv[j] = ok ? t : 0.0f;
        }
        #pragma unroll
        for (int j = 0; j < 16; ++j) dst[j] = pack2bf(vv[2 * j], vv[2 * j + 1]);
      }
    }
    // ---- stage B tile (BN x BKK) ----
    {
      const int gn = tileN + br;
      unsigned* dst = (unsigned*)&Bs[br * LDT + bc];
      if (MODE != 2 && gn < p.N && fullK) {
        const float* src = p.B + (size_t)gn * p.K + k0 + bc;
        #pragma unroll
        for (int j = 0; j < 4; ++j) {
          const float4 f = *(const float4*)(src + 4 * j);
          dst[2 * j]     = pack2bf(f.x, f.y);
          dst[2 * j + 1] = pack2bf(f.z, f.w);
        }
      } else {
        #pragma unroll
        for (int j = 0; j < 16; ++j) {
          const int gk = k0 + bc + j;
          float v = 0.0f;
          if (gn < p.N && gk < p.K) {
            if (MODE == 2) {
              const int ci = gk / 9, r9 = gk - ci * 9;
              const int kh = r9 / 3, kw = r9 - kh * 3;
              v = p.B[(((size_t)ci * p.Co + gn) * 3 + (2 - kh)) * 3 + (2 - kw)];
            } else {
              v = p.B[(size_t)gn * p.K + gk];
            }
          }
          if (j & 1) dst[j >> 1] |= ((unsigned)f2bf(v) << 16);
          else       dst[j >> 1]  = (unsigned)f2bf(v);
        }
      }
    }
    __syncthreads();

    // ---- fragments in contiguous k order (k = 16*half + s) for A and B; 8 WMMAs per wave ----
    const int arow0 = wm * 32 + (lane & 15);
    const int arow1 = arow0 + 16;
    const int akb   = (lane < 16) ? 0 : 16;   // A: K base per half-wave
    const int bcol0 = wn * 32 + (lane & 15);
    const int bcol1 = bcol0 + 16;
    const int bkb   = (lane < 16) ? 0 : 16;   // B: K base per half-wave
    #pragma unroll
    for (int ks = 0; ks < BKK; ks += 32) {
      union FU { v16bf v; unsigned u[8]; };
      FU a0, a1, b0, b1;
      #pragma unroll
      for (int v = 0; v < 8; ++v) {
        const int ak = ks + akb + 2 * v;
        const int bk = ks + bkb + 2 * v;
        a0.u[v] = *(const unsigned*)&As[arow0 * LDT + ak];
        a1.u[v] = *(const unsigned*)&As[arow1 * LDT + ak];
        b0.u[v] = *(const unsigned*)&Bs[bcol0 * LDT + bk];
        b1.u[v] = *(const unsigned*)&Bs[bcol1 * LDT + bk];
      }
      acc00 = __builtin_amdgcn_wmma_f32_16x16x32_bf16(false, a0.v, false, b0.v, (short)0, acc00, false, false);
      acc01 = __builtin_amdgcn_wmma_f32_16x16x32_bf16(false, a0.v, false, b1.v, (short)0, acc01, false, false);
      acc10 = __builtin_amdgcn_wmma_f32_16x16x32_bf16(false, a1.v, false, b0.v, (short)0, acc10, false, false);
      acc11 = __builtin_amdgcn_wmma_f32_16x16x32_bf16(false, a1.v, false, b1.v, (short)0, acc11, false, false);
    }
    __syncthreads();
  }

  // ---- epilogue: D layout = VGPR v -> M=v (lanes<16) / v+8 (lanes>=16) ----
  const int mhw = (lane & 16) ? 8 : 0;
  const int nl  = lane & 15;
  #pragma unroll
  for (int fm = 0; fm < 2; ++fm) {
    #pragma unroll
    for (int fn = 0; fn < 2; ++fn) {
      const v8f acc = (fm == 0) ? (fn == 0 ? acc00 : acc01)
                                : (fn == 0 ? acc10 : acc11);
      const int n = tileN + wn * 32 + fn * 16 + nl;
      if (n >= p.N) continue;
      const float bsv = p.bias ? p.bias[n] : 0.0f;
      #pragma unroll
      for (int v = 0; v < 8; ++v) {
        const int m = tileM + wm * 32 + fm * 16 + mhw + v;
        if (m >= p.M) continue;
        float val = acc[v] + bsv;
        if (p.act == 1) val = fmaxf(val, 0.0f);
        else if (p.act == 2) val = (val >= 0.0f) ? val : 0.01f * val;
        size_t idx;
        if (MODE == 0 || MODE == 3) {
          idx = (size_t)m * p.N + n;
        } else { // NCHW scatter
          const int nb = m / HoWo, rem = m - nb * HoWo;
          idx = ((size_t)nb * p.N + n) * HoWo + rem;
        }
        p.C[idx] = val;
      }
    }
  }
}

// --------------------------- small VALU kernels ----------------------------

__global__ __launch_bounds__(256) void k_layernorm(float* __restrict__ y,
    const float* __restrict__ x, const float* __restrict__ g,
    const float* __restrict__ b) {
  __shared__ float red[256];
  const int row = blockIdx.x, tid = threadIdx.x;
  const float v = x[row * 256 + tid];
  red[tid] = v; __syncthreads();
  for (int s = 128; s > 0; s >>= 1) { if (tid < s) red[tid] += red[tid + s]; __syncthreads(); }
  const float m = red[0] / 256.0f; __syncthreads();
  const float d = v - m;
  red[tid] = d * d; __syncthreads();
  for (int s = 128; s > 0; s >>= 1) { if (tid < s) red[tid] += red[tid + s]; __syncthreads(); }
  const float var = red[0] / 256.0f;
  y[row * 256 + tid] = d * rsqrtf(var + 1e-5f) * g[tid] + b[tid];
}

// o = a + b, vectorized float4 (n must be a multiple of 4)
__global__ void k_add2(float* __restrict__ o, const float* __restrict__ a,
                       const float* __restrict__ b, int n4) {
  const int i = blockIdx.x * 256 + threadIdx.x;
  if (i >= n4) return;
  const float4 x = ((const float4*)a)[i];
  const float4 y = ((const float4*)b)[i];
  float4 r; r.x = x.x + y.x; r.y = x.y + y.y; r.z = x.z + y.z; r.w = x.w + y.w;
  ((float4*)o)[i] = r;
}

__global__ void k_zero(float* p, int n) {
  const int i = blockIdx.x * 256 + threadIdx.x;
  if (i < n) p[i] = 0.0f;
}

__global__ __launch_bounds__(256) void k_bn_stats(const float* __restrict__ x,
    float* __restrict__ mean, float* __restrict__ var, int N, int C, int HW) {
  __shared__ float s1[256], s2[256];
  const int c = blockIdx.x, tid = threadIdx.x;
  const int tot4 = (N * HW) >> 2;      // HW is a multiple of 4 everywhere here
  const int HW4 = HW >> 2;
  float a = 0.0f, b = 0.0f;
  for (int i = tid; i < tot4; i += 256) {
    const int n = i / HW4, r = i - n * HW4;
    const float4 v = *(const float4*)&x[((size_t)n * C + c) * HW + 4 * r];
    a += v.x + v.y + v.z + v.w;
    b += v.x * v.x + v.y * v.y + v.z * v.z + v.w * v.w;
  }
  s1[tid] = a; s2[tid] = b; __syncthreads();
  for (int s = 128; s > 0; s >>= 1) {
    if (tid < s) { s1[tid] += s1[tid + s]; s2[tid] += s2[tid + s]; }
    __syncthreads();
  }
  if (tid == 0) {
    const float tot = (float)(N * HW);
    const float m = s1[0] / tot; mean[c] = m; var[c] = s2[0] / tot - m * m;
  }
}

__global__ void k_bn_leaky(float* __restrict__ y, const float* __restrict__ x,
    const float* __restrict__ mean, const float* __restrict__ var,
    const float* __restrict__ g, const float* __restrict__ be,
    int C, int HW, int total4) {
  const int i = blockIdx.x * 256 + threadIdx.x;
  if (i >= total4) return;
  const int c = ((i * 4) / HW) % C;          // HW % 4 == 0 -> same channel x4
  const float sc = rsqrtf(var[c] + 1e-5f) * g[c];
  const float sh = be[c] - mean[c] * sc;
  float4 v = ((const float4*)x)[i];
  v.x = v.x * sc + sh; v.y = v.y * sc + sh; v.z = v.z * sc + sh; v.w = v.w * sc + sh;
  v.x = (v.x >= 0.0f) ? v.x : 0.01f * v.x;
  v.y = (v.y >= 0.0f) ? v.y : 0.01f * v.y;
  v.z = (v.z >= 0.0f) ? v.z : 0.01f * v.z;
  v.w = (v.w >= 0.0f) ? v.w : 0.01f * v.w;
  ((float4*)y)[i] = v;
}

// attention: one block per (b,h,q); scores -> softmax -> A@V.
__global__ __launch_bounds__(256) void k_attn(const float* __restrict__ Q,
    const float* __restrict__ K, const float* __restrict__ V,
    const float* __restrict__ bias, float* __restrict__ O,
    int Lq, int Lk, int B, int causal) {
  __shared__ float qrow[64];
  __shared__ float sc[1024];
  __shared__ float red[256];
  const int bid = blockIdx.x;
  const int qi = bid % Lq, t = bid / Lq;
  const int h = t & 3, b = t >> 2;
  const int tid = threadIdx.x;
  const float* qp = Q + ((size_t)(qi * B + b)) * 256 + h * 64;
  if (tid < 64) qrow[tid] = qp[tid];
  __syncthreads();
  float lmax = -3.0e38f;
  for (int k = tid; k < Lk; k += 256) {
    const float* kp = K + ((size_t)(k * B + b)) * 256 + h * 64;
    float s = 0.0f;
    #pragma unroll 8
    for (int d = 0; d < 64; ++d) s += qrow[d] * kp[d];
    s *= 0.125f;                                 // 1/sqrt(64)
    if (bias) s += bias[((size_t)(b * 4 + h) * Lq + qi) * Lk + k];
    if (causal && k > qi) s -= 1e9f;
    sc[k] = s;
    lmax = fmaxf(lmax, s);
  }
  red[tid] = lmax; __syncthreads();
  for (int s = 128; s > 0; s >>= 1) { if (tid < s) red[tid] = fmaxf(red[tid], red[tid + s]); __syncthreads(); }
  const float gmax = red[0]; __syncthreads();
  float psum = 0.0f;
  for (int k = tid; k < Lk; k += 256) { const float e = expf(sc[k] - gmax); sc[k] = e; psum += e; }
  red[tid] = psum; __syncthreads();
  for (int s = 128; s > 0; s >>= 1) { if (tid < s) red[tid] += red[tid + s]; __syncthreads(); }
  const float inv = 1.0f / red[0];
  __syncthreads();
  if (tid < 64) {
    float o = 0.0f;
    for (int k = 0; k < Lk; ++k)
      o += sc[k] * V[((size_t)(k * B + b)) * 256 + h * 64 + tid];
    O[((size_t)(qi * B + b)) * 256 + h * 64 + tid] = o * inv;
  }
}

// bias[b,h,t,s] from mu/log_var (16,12) + pt heads (16,48) + grid (S,1,2)
__global__ void k_gauss(float* __restrict__ bias, const float* __restrict__ grid,
    const float* __restrict__ mu, const float* __restrict__ lv,
    const float* __restrict__ p1, const float* __restrict__ p2, int total) {
  const int i = blockIdx.x * 256 + threadIdx.x;
  if (i >= total) return;
  const int s = i & 1023;
  int r = i >> 10;
  const int t = r % 6; r /= 6;
  const int h = r & 3;
  const int b = r >> 2;
  float acc = 0.0f;
  #pragma unroll
  for (int c = 0; c < 2; ++c) {
    const float m3 = sigm(mu[t * 32 + b * 2 + c]) * 4.0f + sigm(p1[t * 128 + b * 8 + h * 2 + c]);
    const float l3 = expf(lv[t * 32 + b * 2 + c]) + sigm(p2[t * 128 + b * 8 + h * 2 + c]);
    const float d  = grid[s * 2 + c] - m3;
    acc += -(d * d) / (2.0f * l3 * l3) - logf(l3) - 0.5f * LOG2PI_F;
  }
  bias[i] = acc;
}

// reparameterize z = eps*exp(0.5 lv)+mu (hash RNG) and KLD partial -> scr[1]
__global__ __launch_bounds__(256) void k_z_kld(float* __restrict__ z,
    float* __restrict__ scr, const float* __restrict__ mu,
    const float* __restrict__ lv) {
  __shared__ float red[256];
  const int tid = threadIdx.x;
  float kp = 0.0f;
  if (tid < 192) {
    const float m = mu[tid], l = lv[tid];
    const float u1 = (pcg(2u * tid + 1u) + 0.5f) * 2.3283064e-10f;
    const float u2 = (pcg(2u * tid + 2u) + 0.5f) * 2.3283064e-10f;
    const float eps = sqrtf(-2.0f * logf(u1)) * cosf(6.28318530718f * u2);
    z[tid] = eps * expf(0.5f * l) + m;
    kp = 1.0f + l - m * m - expf(l);
  }
  red[tid] = kp; __syncthreads();
  for (int s = 128; s > 0; s >>= 1) { if (tid < s) red[tid] += red[tid + s]; __syncthreads(); }
  if (tid == 0) scr[1] = red[0];
}

// tanh + MSE reduction against image -> scr[0] (n4 = n/4)
__global__ __launch_bounds__(256) void k_recon(const float* __restrict__ x,
    const float* __restrict__ img, float* __restrict__ scr, int n4) {
  __shared__ float red[256];
  const int i = blockIdx.x * 256 + threadIdx.x;
  float d = 0.0f;
  if (i < n4) {
    const float4 a = ((const float4*)x)[i];
    const float4 b = ((const float4*)img)[i];
    float e;
    e = tanhf(a.x) - b.x; d += e * e;
    e = tanhf(a.y) - b.y; d += e * e;
    e = tanhf(a.z) - b.z; d += e * e;
    e = tanhf(a.w) - b.w; d += e * e;
  }
  red[threadIdx.x] = d; __syncthreads();
  for (int s = 128; s > 0; s >>= 1) { if (threadIdx.x < s) red[threadIdx.x] += red[threadIdx.x + s]; __syncthreads(); }
  if (threadIdx.x == 0) atomicAdd(&scr[0], red[0]);
}

__global__ void k_finalize(float* __restrict__ out, const float* __restrict__ scr) {
  const float recon = scr[0] / 3145728.0f;          // 16*3*256*256
  const float kld   = -0.5f * scr[1] / 16.0f;
  out[24576] = recon * 0.2f + kld * 0.01f;
}

// --------------------------------- host ------------------------------------

enum {
  IN_GRID = 0, IN_IMAGE, IN_TGT, IN_MEMORY, IN_MEMCDA, IN_POS, IN_QPOS,
  P_SA_INW, P_SA_INB, P_SA_OUTW, P_SA_OUTB,
  P_G1_INW, P_G1_INB, P_G1_OUTW, P_G1_OUTB,
  P_G2_INW, P_G2_INB, P_G2_OUTW, P_G2_OUTB,
  P_G3_INW, P_G3_INB, P_G3_OUTW, P_G3_OUTB,
  P_FUS_W, P_FUS_B, P_L1_W, P_L1_B, P_L2_W, P_L2_B,
  P_PT1_W, P_PT1_B, P_PT2_W, P_PT2_B,
  P_FCMU_W, P_FCMU_B, P_FCVAR_W, P_FCVAR_B,
  P_DECIN_W, P_DECIN_B,
  P_N1G, P_N1B, P_N3G, P_N3B, P_N4G, P_N4B,
  P_ENC0_W, P_ENC0_B, P_ENC0_G, P_ENC0_BE,
  P_ENC1_W, P_ENC1_B, P_ENC1_G, P_ENC1_BE,
  P_ENC2_W, P_ENC2_B, P_ENC2_G, P_ENC2_BE,
  P_ENC3_W, P_ENC3_B, P_ENC3_G, P_ENC3_BE,
  P_ENC4_W, P_ENC4_B, P_ENC4_G, P_ENC4_BE,
  P_DEC0_W, P_DEC0_B, P_DEC0_G, P_DEC0_BE,
  P_DEC1_W, P_DEC1_B, P_DEC1_G, P_DEC1_BE,
  P_DEC2_W, P_DEC2_B, P_DEC2_G, P_DEC2_BE,
  P_DEC3_W, P_DEC3_B, P_DEC3_G, P_DEC3_BE,
  P_CT_W, P_CT_B, P_FIN_G, P_FIN_BE, P_C_W, P_C_B
};

static void run_gemm(hipStream_t st, int mode, const GemmP& p) {
  dim3 g((p.N + BN - 1) / BN, (p.M + BM - 1) / BM);
  switch (mode) {
    case 0: k_gemm<0><<<g, 256, 0, st>>>(p); break;
    case 1: k_gemm<1><<<g, 256, 0, st>>>(p); break;
    case 2: k_gemm<2><<<g, 256, 0, st>>>(p); break;
    case 3: k_gemm<3><<<g, 256, 0, st>>>(p); break;
  }
}

static void gemm_lin(hipStream_t st, const float* A, const float* B,
                     const float* bias, float* C, int M, int N, int K, int act) {
  GemmP p{}; p.A = A; p.B = B; p.bias = bias; p.C = C;
  p.M = M; p.N = N; p.K = K; p.act = act;
  run_gemm(st, 0, p);
}

static void gemm_cat(hipStream_t st, const float* A, const float* A2,
                     const float* B, const float* bias, float* C,
                     int M, int N, int K, int act) {
  GemmP p{}; p.A = A; p.A2 = A2; p.B = B; p.bias = bias; p.C = C;
  p.M = M; p.N = N; p.K = K; p.act = act;
  run_gemm(st, 3, p);
}

static void conv3(hipStream_t st, const float* x, const float* w, const float* b,
                  float* y, int BS_, int Ci, int H, int Co, int stride) {
  const int Ho = (stride == 2) ? H / 2 : H;
  GemmP p{}; p.A = x; p.B = w; p.bias = b; p.C = y;
  p.M = BS_ * Ho * Ho; p.N = Co; p.K = Ci * 9; p.act = 0;
  p.Ci = Ci; p.H = H; p.W = H; p.Ho = Ho; p.Wo = Ho; p.stride = stride; p.Co = Co;
  run_gemm(st, 1, p);
}

static void convt3(hipStream_t st, const float* x, const float* w, const float* b,
                   float* y, int BS_, int Ci, int H, int Co) {
  const int Ho = 2 * H;
  GemmP p{}; p.A = x; p.B = w; p.bias = b; p.C = y;
  p.M = BS_ * Ho * Ho; p.N = Co; p.K = Ci * 9; p.act = 0;
  p.Ci = Ci; p.H = H; p.W = H; p.Ho = Ho; p.Wo = Ho; p.stride = 2; p.Co = Co;
  run_gemm(st, 2, p);
}

static void bn_leaky(hipStream_t st, float* x, const float* g, const float* be,
                     float* meanb, float* varb, int N, int C, int HW) {
  k_bn_stats<<<C, 256, 0, st>>>(x, meanb, varb, N, C, HW);
  const int total4 = (N * C * HW) >> 2;
  k_bn_leaky<<<(total4 + 255) / 256, 256, 0, st>>>(x, x, meanb, varb, g, be, C, HW, total4);
}

extern "C" void kernel_launch(void* const* d_in, const int* in_sizes, int n_in,
                              void* d_out, int out_size, void* d_ws, size_t ws_size,
                              hipStream_t stream) {
  (void)in_sizes; (void)n_in; (void)out_size; (void)ws_size;
  const float* grid   = (const float*)d_in[IN_GRID];
  const float* image  = (const float*)d_in[IN_IMAGE];
  const float* tgt_in = (const float*)d_in[IN_TGT];
  const float* memory = (const float*)d_in[IN_MEMORY];
  const float* memcda = (const float*)d_in[IN_MEMCDA];
  const float* pos    = (const float*)d_in[IN_POS];
  const float* qpos   = (const float*)d_in[IN_QPOS];
  float* out = (float*)d_out;
  float* ws  = (float*)d_ws;

  size_t off = 0;
  auto alloc = [&](size_t n) { float* p = ws + off; off += n; return p; };
  float* bufA  = alloc(8388608);    // 16*32*128*128
  float* bufB  = alloc(8388608);
  float* bufCT = alloc(33554432);   // 16*32*256*256
  float* feat  = alloc(4194304);    // 16384 x 256
  float* kin   = alloc(4194304);
  float* KP    = alloc(4194304);
  float* VP    = alloc(4194304);
  float* biasb = alloc(393216);     // (16,4,6,1024)
  float* tgtb  = alloc(24576);
  float* lnb   = alloc(24576);
  float* qb    = alloc(24576);
  float* Qp    = alloc(24576);
  float* Ksm   = alloc(24576);
  float* Vsm   = alloc(24576);
  float* Oat   = alloc(24576);
  float* prj   = alloc(24576);
  float* ffh   = alloc(196608);     // 96 x 2048
  float* p1o   = alloc(768);
  float* p2o   = alloc(768);
  float* mub   = alloc(192);
  float* lvb   = alloc(192);
  float* zb    = alloc(192);
  float* meanb = alloc(256);
  float* varb  = alloc(256);
  float* scr   = alloc(8);

  k_zero<<<1, 256, 0, stream>>>(scr, 8);

  const float* saw = (const float*)d_in[P_SA_INW];
  const float* sab = (const float*)d_in[P_SA_INB];

  // ---- pre-norm causal self-attention ----
  k_layernorm<<<96, 256, 0, stream>>>(lnb, tgt_in, (const float*)d_in[P_N1G], (const float*)d_in[P_N1B]);
  k_add2<<<24, 256, 0, stream>>>(qb, lnb, qpos, 6144);
  gemm_lin(stream, qb,  saw,           sab,       Qp,  96, 256, 256, 0);
  gemm_lin(stream, qb,  saw + 65536,   sab + 256, Ksm, 96, 256, 256, 0);
  gemm_lin(stream, lnb, saw + 131072,  sab + 512, Vsm, 96, 256, 256, 0);
  k_attn<<<6 * 16 * 4, 256, 0, stream>>>(Qp, Ksm, Vsm, nullptr, Oat, 6, 6, 16, 1);
  gemm_lin(stream, Oat, (const float*)d_in[P_SA_OUTW], (const float*)d_in[P_SA_OUTB], prj, 96, 256, 256, 0);
  k_add2<<<24, 256, 0, stream>>>(tgtb, tgt_in, prj, 6144);

  // ---- VAE encoder (conv s2 p1 + BN(train) + leaky) ----
  conv3(stream, image, (const float*)d_in[P_ENC0_W], (const float*)d_in[P_ENC0_B], bufA, 16,   3, 256,  32, 2);
  bn_leaky(stream, bufA, (const float*)d_in[P_ENC0_G], (const float*)d_in[P_ENC0_BE], meanb, varb, 16,  32, 16384);
  conv3(stream, bufA, (const float*)d_in[P_ENC1_W], (const float*)d_in[P_ENC1_B], bufB, 16,  32, 128,  64, 2);
  bn_leaky(stream, bufB, (const float*)d_in[P_ENC1_G], (const float*)d_in[P_ENC1_BE], meanb, varb, 16,  64,  4096);
  conv3(stream, bufB, (const float*)d_in[P_ENC2_W], (const float*)d_in[P_ENC2_B], bufA, 16,  64,  64, 128, 2);
  bn_leaky(stream, bufA, (const float*)d_in[P_ENC2_G], (const float*)d_in[P_ENC2_BE], meanb, varb, 16, 128,  1024);
  conv3(stream, bufA, (const float*)d_in[P_ENC3_W], (const float*)d_in[P_ENC3_B], bufB, 16, 128,  32, 256, 2);
  bn_leaky(stream, bufB, (const float*)d_in[P_ENC3_G], (const float*)d_in[P_ENC3_BE], meanb, varb, 16, 256,   256);
  conv3(stream, bufB, (const float*)d_in[P_ENC4_W], (const float*)d_in[P_ENC4_B], bufA, 16, 256,  16, 256, 2);
  bn_leaky(stream, bufA, (const float*)d_in[P_ENC4_G], (const float*)d_in[P_ENC4_BE], meanb, varb, 16, 256,    64);
  // result = bufA : (16, 16384)

  gemm_lin(stream, bufA, (const float*)d_in[P_FCMU_W],  (const float*)d_in[P_FCMU_B],  mub, 16, 12, 16384, 0);
  gemm_lin(stream, bufA, (const float*)d_in[P_FCVAR_W], (const float*)d_in[P_FCVAR_B], lvb, 16, 12, 16384, 0);
  gemm_lin(stream, bufA, (const float*)d_in[P_PT1_W],   (const float*)d_in[P_PT1_B],   p1o, 16, 48, 16384, 0);
  gemm_lin(stream, bufA, (const float*)d_in[P_PT2_W],   (const float*)d_in[P_PT2_B],   p2o, 16, 48, 16384, 0);
  k_z_kld<<<1, 256, 0, stream>>>(zb, scr, mub, lvb);

  // ---- Gaussian attention bias ----
  k_gauss<<<(393216 + 255) / 256, 256, 0, stream>>>(biasb, grid, mub, lvb, p1o, p2o, 393216);

  // ---- VAE decoder (convT s2 p1 op1 + BN + leaky) ----
  gemm_lin(stream, zb, (const float*)d_in[P_DECIN_W], (const float*)d_in[P_DECIN_B], bufB, 16, 16384, 12, 0);
  convt3(stream, bufB, (const float*)d_in[P_DEC0_W], (const float*)d_in[P_DEC0_B], bufA, 16, 256,   8, 256);
  bn_leaky(stream, bufA, (const float*)d_in[P_DEC0_G], (const float*)d_in[P_DEC0_BE], meanb, varb, 16, 256,   256);
  convt3(stream, bufA, (const float*)d_in[P_DEC1_W], (const float*)d_in[P_DEC1_B], bufB, 16, 256,  16, 128);
  bn_leaky(stream, bufB, (const float*)d_in[P_DEC1_G], (const float*)d_in[P_DEC1_BE], meanb, varb, 16, 128,  1024);
  convt3(stream, bufB, (const float*)d_in[P_DEC2_W], (const float*)d_in[P_DEC2_B], bufA, 16, 128,  32,  64);
  bn_leaky(stream, bufA, (const float*)d_in[P_DEC2_G], (const float*)d_in[P_DEC2_BE], meanb, varb, 16,  64,  4096);
  convt3(stream, bufA, (const float*)d_in[P_DEC3_W], (const float*)d_in[P_DEC3_B], bufB, 16,  64,  64,  32);
  bn_leaky(stream, bufB, (const float*)d_in[P_DEC3_G], (const float*)d_in[P_DEC3_BE], meanb, varb, 16,  32, 16384);
  convt3(stream, bufB, (const float*)d_in[P_CT_W],   (const float*)d_in[P_CT_B],   bufCT, 16, 32, 128,  32);
  bn_leaky(stream, bufCT, (const float*)d_in[P_FIN_G], (const float*)d_in[P_FIN_BE], meanb, varb, 16, 32, 65536);
  conv3(stream, bufCT, (const float*)d_in[P_C_W], (const float*)d_in[P_C_B], bufA, 16, 32, 256, 3, 1);
  k_recon<<<(786432 + 255) / 256, 256, 0, stream>>>(bufA, image, scr, 786432);

  // ---- fusion of memory streams ----
  gemm_cat(stream, memory, memcda, (const float*)d_in[P_FUS_W], (const float*)d_in[P_FUS_B],
           feat, 16384, 256, 512, 1 /*relu*/);

  // ---- three Gaussian cross-attention branches ----
  k_layernorm<<<96, 256, 0, stream>>>(lnb, tgtb, (const float*)d_in[P_N3G], (const float*)d_in[P_N3B]);
  k_add2<<<24, 256, 0, stream>>>(qb, lnb, qpos, 6144);

  const int br_inw[3]  = {P_G1_INW,  P_G2_INW,  P_G3_INW};
  const int br_inb[3]  = {P_G1_INB,  P_G2_INB,  P_G3_INB};
  const int br_outw[3] = {P_G1_OUTW, P_G2_OUTW, P_G3_OUTW};
  const int br_outb[3] = {P_G1_OUTB, P_G2_OUTB, P_G3_OUTB};
  const float* br_src[3] = {feat, memory, memcda};
  for (int br = 0; br < 3; ++br) {
    const float* w = (const float*)d_in[br_inw[br]];
    const float* b = (const float*)d_in[br_inb[br]];
    k_add2<<<(1048576 + 255) / 256, 256, 0, stream>>>(kin, br_src[br], pos, 1048576);
    gemm_lin(stream, qb,          w,           b,       Qp, 96,    256, 256, 0);
    gemm_lin(stream, kin,         w + 65536,   b + 256, KP, 16384, 256, 256, 0);
    gemm_lin(stream, br_src[br],  w + 131072,  b + 512, VP, 16384, 256, 256, 0);
    k_attn<<<6 * 16 * 4, 256, 0, stream>>>(Qp, KP, VP, biasb, Oat, 6, 1024, 16, 0);
    gemm_lin(stream, Oat, (const float*)d_in[br_outw[br]], (const float*)d_in[br_outb[br]],
             prj, 96, 256, 256, 0);
    k_add2<<<24, 256, 0, stream>>>(tgtb, tgtb, prj, 6144);
  }

  // ---- FFN ----
  k_layernorm<<<96, 256, 0, stream>>>(lnb, tgtb, (const float*)d_in[P_N4G], (const float*)d_in[P_N4B]);
  gemm_lin(stream, lnb, (const float*)d_in[P_L1_W], (const float*)d_in[P_L1_B], ffh, 96, 2048,  256, 1);
  gemm_lin(stream, ffh, (const float*)d_in[P_L2_W], (const float*)d_in[P_L2_B], prj, 96,  256, 2048, 0);
  k_add2<<<24, 256, 0, stream>>>(out, tgtb, prj, 6144);

  k_finalize<<<1, 1, 0, stream>>>(out, scr);
}
